// SSMFoundational_12850542149720
// MI455X (gfx1250) — compile-verified, matrix operand order checked by reading, not run on hardware
//
#include <hip/hip_runtime.h>
#include <hip/hip_bf16.h>
#include <math.h>

// Problem sizes (fixed by the reference)
#define B_   16
#define L_   4096
#define IN_  192
#define H_   512
#define P_   128
#define K_   6
#define OUT_ 192
#define M_   (B_*L_)     // 65536 rows (flattened batch*time)
#define S_   64          // scan chunks per sequence
#define CL_  (L_/S_)     // 64 steps per chunk

typedef __attribute__((ext_vector_type(16))) __bf16 v16bf;
typedef __attribute__((ext_vector_type(8)))  float  v8f;

// ---------------------------------------------------------------------------
// helpers
// ---------------------------------------------------------------------------
__device__ __forceinline__ float gelu_tanh(float x) {
    float x3 = x * x * x;
    return 0.5f * x * (1.0f + tanhf(0.7978845608028654f * (x + 0.044715f * x3)));
}

// Gather one 16x32 bf16 WMMA fragment from an LDS tile stored row-major with
// stride 32 (row = M for A, row = N for B staged transposed).  K pairs are
// contiguous -> compiler lowers this to 2x ds_load_b128 per lane.
__device__ __forceinline__ v16bf frag_ld(const __bf16* rowbase, int half) {
    union { v16bf v; unsigned u[8]; } f;
#pragma unroll
    for (int j = 0; j < 8; ++j) {
        int kk = (j < 4 ? 2 * j : 16 + 2 * (j - 4)) + 8 * half;
        f.u[j] = *(const unsigned*)(rowbase + kk);
    }
    return f.v;
}

// one K-step of compute: gather 4 fragments + 4 WMMAs into acc[2][2]
__device__ __forceinline__ void compute_step(const __bf16* sAbuf, const __bf16* sBbuf,
                                             int wm, int wn, int l16, int half,
                                             v8f acc[2][2]) {
    v16bf fa0 = frag_ld(sAbuf + (wm * 32 +      l16) * 32, half);
    v16bf fa1 = frag_ld(sAbuf + (wm * 32 + 16 + l16) * 32, half);
    v16bf fb0 = frag_ld(sBbuf + (wn * 32 +      l16) * 32, half);
    v16bf fb1 = frag_ld(sBbuf + (wn * 32 + 16 + l16) * 32, half);
    acc[0][0] = __builtin_amdgcn_wmma_f32_16x16x32_bf16(false, fa0, false, fb0,
                                                        (short)0, acc[0][0], false, false);
    acc[0][1] = __builtin_amdgcn_wmma_f32_16x16x32_bf16(false, fa0, false, fb1,
                                                        (short)0, acc[0][1], false, false);
    acc[1][0] = __builtin_amdgcn_wmma_f32_16x16x32_bf16(false, fa1, false, fb0,
                                                        (short)0, acc[1][0], false, false);
    acc[1][1] = __builtin_amdgcn_wmma_f32_16x16x32_bf16(false, fa1, false, fb1,
                                                        (short)0, acc[1][1], false, false);
}

// ---------------------------------------------------------------------------
// Tiled bf16 WMMA GEMM:  C(M,N) = epilogue( A(M,Ka) @ Wt(N,Ka)^T )
// Block tile 128x64, 8 waves, each wave computes a 32x32 output tile
// (4 accumulators, 4 WMMAs per 32-wide K step).  LDS double-buffered with a
// manually 2x-unrolled pipeline: compile-time buffer indices, two independent
// register sets, one barrier per K-step, loads issued a full K-step early.
// Requires nsteps = Ka/32 even and >= 2 (holds for Ka in {192, 256, 512}).
// EPI = 0 : (+ bias if HB)                          -> C
// EPI = 1 : + e0[m,n]*e1[n], gelu                   -> C   (SSM output path)
// EPI = 2 : C = e1[m,n] + e0[m,n]*sigmoid(acc+b)    -> C   (GLU + skip)
// WB      : also store bf16 copy of the result to Cb
// ---------------------------------------------------------------------------
template<int EPI, bool HB, bool WB>
__global__ __launch_bounds__(256)
void wmma_gemm(const __bf16* __restrict__ A, const __bf16* __restrict__ Wt,
               const float* __restrict__ bias, float* __restrict__ C,
               __bf16* __restrict__ Cb,
               int M, int N, int Ka,
               const float* __restrict__ e0, const float* __restrict__ e1) {
    __shared__ __bf16 sA[2][128 * 32];  // [buf][m][k]
    __shared__ __bf16 sB[2][64 * 32];   // [buf][n][k]  (B staged transposed)

    const int tid  = threadIdx.x;
    const int wv   = tid >> 5;
    const int lane = tid & 31;
    const int wm   = wv >> 1;        // 4 wave-rows  -> 32-row slabs
    const int wn   = wv & 1;         // 2 wave-cols  -> 32-col slabs
    const long m0  = (long)blockIdx.y * 128;
    const int  n0  = blockIdx.x * 64;
    const int  half = lane >> 4;
    const int  l16  = lane & 15;

    // uint4 tile-copy coordinates (8 bf16 per uint4, 4 uint4 per 32-wide row)
    const int arow0 = (tid + 0)   >> 2, acol0 = (tid + 0)   & 3;
    const int arow1 = (tid + 256) >> 2, acol1 = (tid + 256) & 3;
    const int brow  = tid >> 2,         bcol  = tid & 3;

    const __bf16* aptr0 = A  + (m0 + arow0) * (long)Ka + acol0 * 8;
    const __bf16* aptr1 = A  + (m0 + arow1) * (long)Ka + acol1 * 8;
    const __bf16* bptr  = Wt + (long)(n0 + brow) * Ka + bcol * 8;

    v8f acc[2][2] = {{{}, {}}, {{}, {}}};
    uint4 a00, a01, b0;   // register set for even K-steps (buffer 0)
    uint4 a10, a11, b1;   // register set for odd  K-steps (buffer 1)

    const int nsteps = Ka >> 5;       // even, >= 2 for all our GEMMs

    // prologue: load + stage K-step 0; preload K-step 1
    a00 = *(const uint4*)(aptr0);
    a01 = *(const uint4*)(aptr1);
    b0  = *(const uint4*)(bptr);
    ((uint4*)sA[0])[tid]       = a00;
    ((uint4*)sA[0])[tid + 256] = a01;
    ((uint4*)sB[0])[tid]       = b0;
    a10 = *(const uint4*)(aptr0 + 32);
    a11 = *(const uint4*)(aptr1 + 32);
    b1  = *(const uint4*)(bptr  + 32);
    __syncthreads();

    for (int it = 0; it < nsteps; it += 2) {
        // ---- even half: compute buffer 0 (step it) ----
        ((uint4*)sA[1])[tid]       = a10;          // stage step it+1 -> buf 1
        ((uint4*)sA[1])[tid + 256] = a11;
        ((uint4*)sB[1])[tid]       = b1;
        if (it + 2 < nsteps) {                     // preload step it+2
            int kk = (it + 2) << 5;
            a00 = *(const uint4*)(aptr0 + kk);
            a01 = *(const uint4*)(aptr1 + kk);
            b0  = *(const uint4*)(bptr  + kk);
        }
        compute_step(sA[0], sB[0], wm, wn, l16, half, acc);
        __syncthreads();                           // buf 1 ready

        // ---- odd half: compute buffer 1 (step it+1) ----
        if (it + 2 < nsteps) {                     // stage step it+2 -> buf 0
            ((uint4*)sA[0])[tid]       = a00;
            ((uint4*)sA[0])[tid + 256] = a01;
            ((uint4*)sB[0])[tid]       = b0;
        }
        if (it + 3 < nsteps) {                     // preload step it+3
            int kk = (it + 3) << 5;
            a10 = *(const uint4*)(aptr0 + kk);
            a11 = *(const uint4*)(aptr1 + kk);
            b1  = *(const uint4*)(bptr  + kk);
        }
        compute_step(sA[1], sB[1], wm, wn, l16, half, acc);
        __syncthreads();                           // buf 0 ready
    }

    // epilogue: C/D layout -> VGPR r holds (m = r + 8*half, n = lane%16)
#pragma unroll
    for (int mt = 0; mt < 2; ++mt) {
#pragma unroll
        for (int r = 0; r < 8; ++r) {
            long m = m0 + wm * 32 + mt * 16 + r + 8 * half;
#pragma unroll
            for (int nt = 0; nt < 2; ++nt) {
                int n = n0 + wn * 32 + nt * 16 + l16;
                float v = acc[mt][nt][r];
                long  o = m * N + n;
                float res;
                if (EPI == 0) {
                    if (HB) v += bias[n];
                    res = v;
                } else if (EPI == 1) {
                    v += e0[o] * e1[n];               // + u * D
                    res = gelu_tanh(v);
                } else {
                    v += bias[n];
                    float sg = 1.0f / (1.0f + expf(-v));
                    res = e1[o] + e0[o] * sg;         // skip + gelu_y * sigmoid
                }
                C[o] = res;
                if (WB) Cb[o] = (__bf16)res;
            }
        }
    }
}

// ---------------------------------------------------------------------------
// f32 -> bf16 bulk convert (4 elems/thread, 8-byte stores)
// ---------------------------------------------------------------------------
__global__ void f32_to_bf16_k(const float* __restrict__ s, __bf16* __restrict__ d, long n) {
    long i = ((long)blockIdx.x * blockDim.x + threadIdx.x) * 4;
    if (i >= n) return;
    float4 v = *(const float4*)(s + i);
    union { __bf16 b[4]; uint2 u; } o;
    o.b[0] = (__bf16)v.x; o.b[1] = (__bf16)v.y;
    o.b[2] = (__bf16)v.z; o.b[3] = (__bf16)v.w;
    *(uint2*)(d + i) = o.u;
}

// ---------------------------------------------------------------------------
// Per-layer weight prep (bf16 outputs): ZOH discretization + concat weights
//   Bbar (2P x H):  rows [0,P)=Re, [P,2P)=Im of ((lam_bar-1)/lam) * Bm
//   Ccat (H x 2P):  cols [0,P)=2*C_re, [P,2P)=-2*C_im
//   lam  (2P f32):  lam_bar re | im
// ---------------------------------------------------------------------------
__global__ void prep_layer_k(const float* __restrict__ lre, const float* __restrict__ lim,
                             const float* __restrict__ lstep,
                             const float* __restrict__ bre, const float* __restrict__ bim,
                             const float* __restrict__ cre, const float* __restrict__ cim,
                             __bf16* __restrict__ Bbar, __bf16* __restrict__ Ccat,
                             float* __restrict__ lam) {
    int t = blockIdx.x * blockDim.x + threadIdx.x;
    if (t >= P_ * H_) return;
    int p = t / H_, h = t % H_;

    float lr = lre[p], li = lim[p];
    float st = expf(lstep[p]);
    float er = expf(lr * st);
    float abr = er * cosf(li * st);
    float abi = er * sinf(li * st);
    float nr = abr - 1.0f, ni = abi;
    float den = lr * lr + li * li;
    float cr = (nr * lr + ni * li) / den;
    float ci = (ni * lr - nr * li) / den;

    float br = bre[p * H_ + h], bi = bim[p * H_ + h];
    Bbar[p * H_ + h]        = (__bf16)(cr * br - ci * bi);
    Bbar[(P_ + p) * H_ + h] = (__bf16)(cr * bi + ci * br);

    Ccat[h * 2 * P_ + p]      = (__bf16)( 2.0f * cre[h * P_ + p]);
    Ccat[h * 2 * P_ + P_ + p] = (__bf16)(-2.0f * cim[h * P_ + p]);

    if (h == 0) { lam[p] = abr; lam[P_ + p] = abi; }
}

// ---------------------------------------------------------------------------
// LayerNorm: one wave32 per row of (M, H); writes f32 + bf16 mirror
// ---------------------------------------------------------------------------
__global__ __launch_bounds__(256)
void layernorm_k(const float* __restrict__ x, float* __restrict__ y,
                 __bf16* __restrict__ yb,
                 const float* __restrict__ sc, const float* __restrict__ bi) {
    int wave = threadIdx.x >> 5, lane = threadIdx.x & 31;
    long row = (long)blockIdx.x * 8 + wave;
    const float* xr = x + row * H_;
    float v[16];
    float s = 0.f, s2 = 0.f;
#pragma unroll
    for (int i = 0; i < 16; ++i) {
        v[i] = xr[lane + 32 * i];
        s += v[i]; s2 += v[i] * v[i];
    }
#pragma unroll
    for (int o = 16; o; o >>= 1) {
        s  += __shfl_xor(s,  o, 32);
        s2 += __shfl_xor(s2, o, 32);
    }
    float mu  = s  * (1.0f / H_);
    float var = s2 * (1.0f / H_) - mu * mu;
    float rs  = rsqrtf(var + 1e-5f);
    float* yr = y + row * H_;
    __bf16* ybr = yb + row * H_;
#pragma unroll
    for (int i = 0; i < 16; ++i) {
        int n = lane + 32 * i;
        float val = (v[i] - mu) * rs * sc[n] + bi[n];
        yr[n]  = val;
        ybr[n] = (__bf16)val;
    }
}

// ---------------------------------------------------------------------------
// 3-phase chunked complex scan over L (x_t = lam_bar * x_{t-1} + Bu_t).
// Bu layout: (B, L, 2P) f32, re in [0,P), im in [P,2P).
// Phase 3 writes the bf16 state sequence consumed by the output GEMM.
// ---------------------------------------------------------------------------
__global__ void scan_phase1(const float* __restrict__ bu,
                            float* __restrict__ sxr, float* __restrict__ sxi,
                            const float* __restrict__ lam) {
    int idx = blockIdx.x * blockDim.x + threadIdx.x;      // B*P*S threads
    int p = idx % P_;
    int c = (idx / P_) % S_;
    int b = idx / (P_ * S_);
    float ar = lam[p], ai = lam[P_ + p];
    float xr = 0.f, xi = 0.f;
    long base = ((long)b * L_ + (long)c * CL_) * (2 * P_);
    for (int t = 0; t < CL_; ++t) {
        float br = bu[base + t * 2 * P_ + p];
        float bi = bu[base + t * 2 * P_ + P_ + p];
        float nxr = ar * xr - ai * xi + br;
        float nxi = ar * xi + ai * xr + bi;
        xr = nxr; xi = nxi;
    }
    long o = ((long)b * S_ + c) * P_ + p;
    sxr[o] = xr; sxi[o] = xi;
}

__global__ void scan_phase2(float* __restrict__ sxr, float* __restrict__ sxi,
                            const float* __restrict__ lam) {
    int idx = blockIdx.x * blockDim.x + threadIdx.x;      // B*P threads
    if (idx >= B_ * P_) return;
    int p = idx % P_, b = idx / P_;
    float ar = lam[p], ai = lam[P_ + p];
    float pr = 1.f, pi = 0.f;                             // lam_bar^CL
    for (int i = 0; i < CL_; ++i) {
        float nr = pr * ar - pi * ai;
        pi = pr * ai + pi * ar;
        pr = nr;
    }
    float cr = 0.f, ci = 0.f;                             // carry into chunk 0
    for (int c = 0; c < S_; ++c) {
        long o = ((long)b * S_ + c) * P_ + p;
        float oxr = sxr[o], oxi = sxi[o];
        sxr[o] = cr; sxi[o] = ci;
        float nr = pr * cr - pi * ci + oxr;
        float ni = pr * ci + pi * cr + oxi;
        cr = nr; ci = ni;
    }
}

__global__ void scan_phase3(const float* __restrict__ bu, __bf16* __restrict__ bub,
                            const float* __restrict__ sxr, const float* __restrict__ sxi,
                            const float* __restrict__ lam) {
    int idx = blockIdx.x * blockDim.x + threadIdx.x;      // B*P*S threads
    int p = idx % P_;
    int c = (idx / P_) % S_;
    int b = idx / (P_ * S_);
    float ar = lam[p], ai = lam[P_ + p];
    long so = ((long)b * S_ + c) * P_ + p;
    float xr = sxr[so], xi = sxi[so];
    long base = ((long)b * L_ + (long)c * CL_) * (2 * P_);
    for (int t = 0; t < CL_; ++t) {
        float br = bu[base + t * 2 * P_ + p];
        float bi = bu[base + t * 2 * P_ + P_ + p];
        float nxr = ar * xr - ai * xi + br;
        float nxi = ar * xi + ai * xr + bi;
        xr = nxr; xi = nxi;
        bub[base + t * 2 * P_ + p]      = (__bf16)xr;
        bub[base + t * 2 * P_ + P_ + p] = (__bf16)xi;
    }
}

// ---------------------------------------------------------------------------
// Host orchestration
// ---------------------------------------------------------------------------
extern "C" void kernel_launch(void* const* d_in, const int* in_sizes, int n_in,
                              void* d_out, int out_size, void* d_ws, size_t ws_size,
                              hipStream_t stream) {
    const float* x      = (const float*)d_in[0];
    const float* W_enc  = (const float*)d_in[1];
    const float* b_enc  = (const float*)d_in[2];
    const float* ln_s   = (const float*)d_in[3];
    const float* ln_b   = (const float*)d_in[4];
    const float* Lre    = (const float*)d_in[5];
    const float* Lim    = (const float*)d_in[6];
    const float* Bre    = (const float*)d_in[7];
    const float* Bim    = (const float*)d_in[8];
    const float* Cre    = (const float*)d_in[9];
    const float* Cim    = (const float*)d_in[10];
    const float* Dv     = (const float*)d_in[11];
    const float* lstep  = (const float*)d_in[12];
    const float* Wglu   = (const float*)d_in[13];
    const float* bglu   = (const float*)d_in[14];
    const float* W_dec  = (const float*)d_in[15];
    const float* b_dec  = (const float*)d_in[16];
    float* out = (float*)d_out;

    // workspace carving (256B-aligned regions)
    char* wsp = (char*)d_ws;
    auto carve = [&](size_t bytes) -> void* {
        void* p = (void*)wsp;
        wsp += (bytes + 255) & ~(size_t)255;
        return p;
    };
    float*  h     = (float*) carve((size_t)M_ * H_ * 4);
    float*  tmp   = (float*) carve((size_t)M_ * H_ * 4);
    float*  bu    = (float*) carve((size_t)M_ * 2 * P_ * 4);
    __bf16* hb    = (__bf16*)carve((size_t)M_ * H_ * 2);
    __bf16* tmpb  = (__bf16*)carve((size_t)M_ * H_ * 2);
    __bf16* bub   = (__bf16*)carve((size_t)M_ * 2 * P_ * 2);
    __bf16* xb    = (__bf16*)carve((size_t)M_ * IN_ * 2);
    __bf16* WencB = (__bf16*)carve((size_t)H_ * IN_ * 2);
    __bf16* WdecB = (__bf16*)carve((size_t)OUT_ * H_ * 2);
    __bf16* WgluB = (__bf16*)carve((size_t)K_ * H_ * H_ * 2);
    __bf16* Bbar  = (__bf16*)carve((size_t)2 * P_ * H_ * 2);
    __bf16* Ccat  = (__bf16*)carve((size_t)H_ * 2 * P_ * 2);
    float*  lam   = (float*) carve((size_t)2 * P_ * 4);
    float*  sxr   = (float*) carve((size_t)B_ * S_ * P_ * 4);
    float*  sxi   = (float*) carve((size_t)B_ * S_ * P_ * 4);

    dim3 blk(256);
    auto cvt = [&](const float* s, __bf16* d, long n) {
        f32_to_bf16_k<<<(unsigned)((n / 4 + 255) / 256), blk, 0, stream>>>(s, d, n);
    };

    // one-time bf16 conversions (input + static weights)
    cvt(x,     xb,    (long)M_ * IN_);
    cvt(W_enc, WencB, (long)H_ * IN_);
    cvt(Wglu,  WgluB, (long)K_ * H_ * H_);
    cvt(W_dec, WdecB, (long)OUT_ * H_);

    // encoder: h = x @ W_enc.T + b_enc
    wmma_gemm<0, true, false><<<dim3(H_ / 64, M_ / 128), blk, 0, stream>>>(
        xb, WencB, b_enc, h, nullptr, M_, H_, IN_, nullptr, nullptr);

    for (int k = 0; k < K_; ++k) {
        prep_layer_k<<<(P_ * H_ + 255) / 256, blk, 0, stream>>>(
            Lre + k * P_, Lim + k * P_, lstep + k * P_,
            Bre + (size_t)k * P_ * H_, Bim + (size_t)k * P_ * H_,
            Cre + (size_t)k * H_ * P_, Cim + (size_t)k * H_ * P_,
            Bbar, Ccat, lam);

        // prenorm -> tmp (f32) + tmpb (bf16)
        layernorm_k<<<M_ / 8, blk, 0, stream>>>(h, tmp, tmpb,
                                                ln_s + k * H_, ln_b + k * H_);

        // Bu = ln_h @ Bbar.T -> (M, 2P) f32
        wmma_gemm<0, false, false><<<dim3(2 * P_ / 64, M_ / 128), blk, 0, stream>>>(
            tmpb, Bbar, nullptr, bu, nullptr, M_, 2 * P_, H_, nullptr, nullptr);

        // linear recurrence over L (chunked 3-phase scan) -> bub (bf16)
        scan_phase1<<<(B_ * P_ * S_) / 256, blk, 0, stream>>>(bu, sxr, sxi, lam);
        scan_phase2<<<(B_ * P_ + 255) / 256, blk, 0, stream>>>(sxr, sxi, lam);
        scan_phase3<<<(B_ * P_ * S_) / 256, blk, 0, stream>>>(bu, bub, sxr, sxi, lam);

        // y = xs @ Ccat.T + u*D ; gelu -> tmp (f32) + tmpb (bf16)
        wmma_gemm<1, false, true><<<dim3(H_ / 64, M_ / 128), blk, 0, stream>>>(
            bub, Ccat, nullptr, tmp, tmpb, M_, H_, 2 * P_, tmp, Dv + k * H_);

        // h = skip + gelu_y * sigmoid(gelu_y @ W_glu.T + b_glu) -> h + hb
        wmma_gemm<2, true, true><<<dim3(H_ / 64, M_ / 128), blk, 0, stream>>>(
            tmpb, WgluB + (size_t)k * H_ * H_, bglu + k * H_, h, hb, M_, H_, H_,
            tmp, h);
    }

    // decoder: out = h @ W_dec.T + b_dec
    wmma_gemm<0, true, false><<<dim3(OUT_ / 64, M_ / 128), blk, 0, stream>>>(
        hb, WdecB, b_dec, out, nullptr, M_, OUT_, H_, nullptr, nullptr);
}
